// GAT_14328010900293
// MI455X (gfx1250) — matrix operand order repaired and checked by reading, not compiled, not run on hardware
//
#include <hip/hip_runtime.h>
#include <math.h>

typedef __attribute__((ext_vector_type(16))) _Float16 v16h;
typedef __attribute__((ext_vector_type(8)))  float    v8f;
typedef unsigned int uint32;

#define HEADS 4
#define HID   64
#define CHN   256      // HEADS * HID
#define TPB   256
#define BN_EPS 1e-5f

static inline int cdiv_i(int a, int b) { return (a + b - 1) / b; }

// ---------------------------------------------------------------------------
// CDNA5 async global->LDS copy (16B per lane), tracked by ASYNCcnt.
__device__ __forceinline__ void async_g2l_b128(uint32 lds_byte_off, const void* gptr) {
  unsigned long long ga = (unsigned long long)(uintptr_t)gptr;
  asm volatile("global_load_async_to_lds_b128 %0, %1, off"
               :: "v"(lds_byte_off), "v"(ga) : "memory");
}
__device__ __forceinline__ void wait_async0() {
  asm volatile("s_wait_asynccnt 0x0" ::: "memory");
}

// ---------------------------------------------------------------------------
// f32 -> f16 convert
__global__ void k_cvt16(const float* __restrict__ src, _Float16* __restrict__ dst, int n) {
  int gid = blockIdx.x * blockDim.x + threadIdx.x;
  if (gid < n) dst[gid] = (_Float16)src[gid];
}

// f32 W[din][dout] -> f16 W^T[dout][din]
__global__ void k_cvtT16(const float* __restrict__ W, _Float16* __restrict__ WT,
                         int din, int dout) {
  int gid = blockIdx.x * blockDim.x + threadIdx.x;
  if (gid >= din * dout) return;
  int r = gid / dout, c = gid - r * dout;
  WT[(size_t)c * din + r] = (_Float16)W[gid];
}

__global__ void k_fill(float* __restrict__ p, float v, int n) {
  int gid = blockIdx.x * blockDim.x + threadIdx.x;
  if (gid < n) p[gid] = v;
}

// ---------------------------------------------------------------------------
// WMMA GEMM: C[M][NOUT] = A16[M][K] * B, with B given as BT16[NOUT][K].
// Workgroup = 8 waves; tile 128(M) x 64(N), K step 32.
// Double-buffered software pipeline: async global->LDS copies (ASYNCcnt) for
// tile k+1 are in flight while tile k's WMMAs execute from the other buffer.
// Out-of-range M rows are clamped (row m of C depends only on row m of A;
// stores are guarded).
__global__ __launch_bounds__(256) void k_gemm_f16_wmma(
    const _Float16* __restrict__ A, const _Float16* __restrict__ BT,
    float* __restrict__ C, int M, int K, int NOUT)
{
  constexpr int BM = 128, BN = 64, BK = 32;
  constexpr int STR = 48;                       // halves/row = 96B (16B-aligned chunks)
  __shared__ _Float16 Ast[2][BM * STR];
  __shared__ _Float16 Bst[2][BN * STR];

  const int t    = threadIdx.x;
  const int lane = t & 31;
  const int wid  = t >> 5;                      // 0..7, one 16-row strip each
  const int m0   = blockIdx.x * BM;
  const int n0   = blockIdx.y * BN;
  const int frow = lane & 15;                   // A row / B col within fragment
  const int kbA  = (lane >> 4) << 3;            // A: halves 0 or 8  (+16 for upper dwords)
  const int kbB  = (lane >> 4) << 3;            // B: halves 0 or 8  (+16 for upper dwords)

  // Cooperative copy assignments: 16B chunks (8 halves).
  // A tile: 128 rows x 4 chunks = 512 chunks -> 2 per thread.
  const int rA0 = t >> 2,       cA = t & 3;     // rows 0..63
  const int rA1 = rA0 + 64;                     // rows 64..127
  int gA0 = m0 + rA0; if (gA0 >= M) gA0 = M - 1;
  int gA1 = m0 + rA1; if (gA1 >= M) gA1 = M - 1;
  // B tile: 64 rows x 4 chunks = 256 chunks -> 1 per thread.
  const int rB = t >> 2, cB = t & 3;

  uint32 ldsA0[2], ldsA1[2], ldsB[2];
  #pragma unroll
  for (int b = 0; b < 2; b++) {
    ldsA0[b] = (uint32)(uintptr_t)&Ast[b][rA0 * STR + cA * 8];
    ldsA1[b] = (uint32)(uintptr_t)&Ast[b][rA1 * STR + cA * 8];
    ldsB[b]  = (uint32)(uintptr_t)&Bst[b][rB  * STR + cB * 8];
  }
  const _Float16* gpA0 = &A[(size_t)gA0 * K + cA * 8];
  const _Float16* gpA1 = &A[(size_t)gA1 * K + cA * 8];
  const _Float16* gpB  = &BT[(size_t)(n0 + rB) * K + cB * 8];

  v8f acc[4] = {};                              // 4 N-subtiles of 16x16 f32

  // pipeline prologue: stage tile k0=0 into buffer 0
  async_g2l_b128(ldsA0[0], gpA0);
  async_g2l_b128(ldsA1[0], gpA1);
  async_g2l_b128(ldsB[0],  gpB);

  int buf = 0;
  for (int k0 = 0; k0 < K; k0 += BK) {
    wait_async0();                              // current tile's copies done (this wave)
    __syncthreads();                            // ... and every other wave's

    const int k1 = k0 + BK;
    if (k1 < K) {                               // kick off next tile into other buffer
      if (k1 + BK < K)                          // gfx1250 global_prefetch_b8 (k+2)
        __builtin_prefetch(gpA0 + k1 + BK, 0, 0);
      async_g2l_b128(ldsA0[buf ^ 1], gpA0 + k1);
      async_g2l_b128(ldsA1[buf ^ 1], gpA1 + k1);
      async_g2l_b128(ldsB[buf ^ 1],  gpB  + k1);
    }

    union { v16h v; uint32 u[8]; } af, bf;
    // A fragment per ISA 16-bit A layout: lanes 0-15 K={0..7,16..23}, lanes 16-31 +8
    const uint32* ap = (const uint32*)&Ast[buf][(wid * 16 + frow) * STR];
    #pragma unroll
    for (int i = 0; i < 4; i++) {
      af.u[i]     = ap[(kbA >> 1) + i];
      af.u[4 + i] = ap[((kbA + 16) >> 1) + i];
    }
    #pragma unroll
    for (int nt = 0; nt < 4; nt++) {
      // B fragment matching A: lanes 0-15 K={0..7,16..23}, lanes 16-31 +8
      const uint32* bp = (const uint32*)&Bst[buf][(nt * 16 + frow) * STR];
      #pragma unroll
      for (int i = 0; i < 4; i++) {
        bf.u[i]     = bp[(kbB >> 1) + i];
        bf.u[4 + i] = bp[((kbB + 16) >> 1) + i];
      }
      acc[nt] = __builtin_amdgcn_wmma_f32_16x16x32_f16(
          false, af.v, false, bf.v, (short)0, acc[nt], false, false);
    }
    buf ^= 1;
    // No trailing barrier needed: each wave's ds_loads for this tile complete
    // (s_wait_dscnt) before its WMMAs issue, and the barrier at the top of the
    // next iteration orders buffer reuse across waves.
  }

  // C/D layout: VGPR r -> M = r (lanes 0-15) or 8+r (lanes 16-31); N = lane&15
  const int mbase = m0 + wid * 16 + ((lane >> 4) << 3);
  #pragma unroll
  for (int nt = 0; nt < 4; nt++) {
    int col = n0 + nt * 16 + (lane & 15);
    #pragma unroll
    for (int r = 0; r < 8; r++) {
      int row = mbase + r;
      if (row < M) C[(size_t)row * NOUT + col] = acc[nt][r];
    }
  }
}

// ---------------------------------------------------------------------------
// a_src[n,h] = sum_c hw[n,h*64+c]*att_src[h,c] ; same for a_dst
__global__ void k_att(const float* __restrict__ hw,
                      const float* __restrict__ atts, const float* __restrict__ attd,
                      float* __restrict__ as_, float* __restrict__ ad_, int n) {
  int gid = blockIdx.x * blockDim.x + threadIdx.x;
  if (gid >= n) return;
  int node = gid >> 2, h = gid & 3;
  const float* hp = hw + (size_t)node * CHN + h * HID;
  const float* ws = atts + h * HID;
  const float* wd = attd + h * HID;
  float ss = 0.f, sd = 0.f;
  #pragma unroll 8
  for (int c = 0; c < HID; c++) { float v = hp[c]; ss += v * ws[c]; sd += v * wd[c]; }
  as_[gid] = ss; ad_[gid] = sd;
}

__device__ inline float leaky02(float x) { return x > 0.f ? x : 0.2f * x; }

__global__ void k_edge_max(const int* __restrict__ ei, int E_, int Et,
                           const float* __restrict__ as_, const float* __restrict__ ad_,
                           float* __restrict__ mx, int n) {
  int gid = blockIdx.x * blockDim.x + threadIdx.x;
  if (gid >= n) return;
  int e = gid >> 2, h = gid & 3;
  int s, d;
  if (e < E_) { s = ei[e]; d = ei[E_ + e]; } else { s = d = e - E_; }
  float lg = leaky02(as_[s * 4 + h] + ad_[d * 4 + h]);
  atomicMax(&mx[d * 4 + h], lg);                 // global_atomic_max_num_f32
}

__global__ void k_edge_denom(const int* __restrict__ ei, int E_, int Et,
                             const float* __restrict__ as_, const float* __restrict__ ad_,
                             const float* __restrict__ mx, float* __restrict__ dn, int n) {
  int gid = blockIdx.x * blockDim.x + threadIdx.x;
  if (gid >= n) return;
  int e = gid >> 2, h = gid & 3;
  int s, d;
  if (e < E_) { s = ei[e]; d = ei[E_ + e]; } else { s = d = e - E_; }
  float lg = leaky02(as_[s * 4 + h] + ad_[d * 4 + h]);
  atomicAdd(&dn[d * 4 + h], __expf(lg - mx[d * 4 + h]));
}

// one wave32 per edge: 256 channels, 8 per lane (coalesced ch = lane + 32*j)
__global__ void k_edge_aggr(const int* __restrict__ ei, int E_, int Et,
                            const float* __restrict__ as_, const float* __restrict__ ad_,
                            const float* __restrict__ mx, const float* __restrict__ dn,
                            const float* __restrict__ hw, float* __restrict__ acc) {
  int gid = blockIdx.x * blockDim.x + threadIdx.x;
  int e = gid >> 5, lane = gid & 31;
  if (e >= Et) return;
  int s, d;
  if (e < E_) { s = ei[e]; d = ei[E_ + e]; } else { s = d = e - E_; }
  float alpha[HEADS];
  #pragma unroll
  for (int h = 0; h < HEADS; h++) {
    float lg = leaky02(as_[s * 4 + h] + ad_[d * 4 + h]);
    alpha[h] = __expf(lg - mx[d * 4 + h]) / (dn[d * 4 + h] + 1e-16f);
  }
  const float* hs = hw  + (size_t)s * CHN;
  float*       od = acc + (size_t)d * CHN;
  #pragma unroll
  for (int j = 0; j < 8; j++) {
    int ch = lane + (j << 5);
    atomicAdd(&od[ch], hs[ch] * alpha[ch >> 6]);
  }
}

// layers 0/1: bias + BN(eval) + ELU, emit f16 for next layer's WMMA GEMM
__global__ void k_epilogue(const float* __restrict__ acc, const float* __restrict__ bias,
                           const float* __restrict__ g, const float* __restrict__ bt,
                           const float* __restrict__ mn, const float* __restrict__ vr,
                           _Float16* __restrict__ out16, int n) {
  int gid = blockIdx.x * blockDim.x + threadIdx.x;
  if (gid >= n) return;
  int ch = gid & (CHN - 1);
  float v = acc[gid] + bias[ch];
  v = (v - mn[ch]) * (g[ch] * rsqrtf(vr[ch] + BN_EPS)) + bt[ch];
  v = v > 0.f ? v : (__expf(v) - 1.f);
  out16[gid] = (_Float16)v;
}

// layer 2: mean over heads + bias + BN (no ELU)
__global__ void k_final2(const float* __restrict__ acc, const float* __restrict__ bias,
                         const float* __restrict__ g, const float* __restrict__ bt,
                         const float* __restrict__ mn, const float* __restrict__ vr,
                         float* __restrict__ h2, int n) {
  int gid = blockIdx.x * blockDim.x + threadIdx.x;
  if (gid >= n) return;
  int node = gid >> 6, c = gid & 63;
  const float* a = acc + (size_t)node * CHN;
  float v = 0.25f * (a[c] + a[64 + c] + a[128 + c] + a[192 + c]) + bias[c];
  v = (v - mn[c]) * (g[c] * rsqrtf(vr[c] + BN_EPS)) + bt[c];
  h2[gid] = v;
}

// classifier MLP: 64 -> 32 (ELU) -> 40
__global__ void k_cls(const float* __restrict__ h2,
                      const float* __restrict__ W0, const float* __restrict__ b0,
                      const float* __restrict__ W1, const float* __restrict__ b1,
                      float* __restrict__ out, int n) {
  int node = blockIdx.x * blockDim.x + threadIdx.x;
  if (node >= n) return;
  float hb[64];
  #pragma unroll 8
  for (int c = 0; c < 64; c++) hb[c] = h2[(size_t)node * 64 + c];
  float t1[32];
  for (int j = 0; j < 32; j++) {
    float s = b0[j];
    #pragma unroll 8
    for (int c = 0; c < 64; c++) s += hb[c] * W0[c * 32 + j];
    t1[j] = s > 0.f ? s : (__expf(s) - 1.f);
  }
  for (int o = 0; o < 40; o++) {
    float s = b1[o];
    #pragma unroll 8
    for (int j = 0; j < 32; j++) s += t1[j] * W1[j * 40 + o];
    out[(size_t)node * 40 + o] = s;
  }
}

// ---------------------------------------------------------------------------
extern "C" void kernel_launch(void* const* d_in, const int* in_sizes, int n_in,
                              void* d_out, int out_size, void* d_ws, size_t ws_size,
                              hipStream_t stream) {
  const float* x  = (const float*)d_in[0];
  const int*   ei = (const int*)d_in[1];
  const int N  = in_sizes[0] / 128;
  const int E  = in_sizes[1] / 2;
  const int Et = E + N;                           // + self loops

  const float* Wl[3] = {(const float*)d_in[2],  (const float*)d_in[6],  (const float*)d_in[10]};
  const float* As[3] = {(const float*)d_in[3],  (const float*)d_in[7],  (const float*)d_in[11]};
  const float* Ad[3] = {(const float*)d_in[4],  (const float*)d_in[8],  (const float*)d_in[12]};
  const float* Bi[3] = {(const float*)d_in[5],  (const float*)d_in[9],  (const float*)d_in[13]};
  const float* Bg[3] = {(const float*)d_in[14], (const float*)d_in[18], (const float*)d_in[22]};
  const float* Bb[3] = {(const float*)d_in[15], (const float*)d_in[19], (const float*)d_in[23]};
  const float* Bm[3] = {(const float*)d_in[16], (const float*)d_in[20], (const float*)d_in[24]};
  const float* Bv[3] = {(const float*)d_in[17], (const float*)d_in[21], (const float*)d_in[25]};
  const float* cW0 = (const float*)d_in[26];
  const float* cb0 = (const float*)d_in[27];
  const float* cW1 = (const float*)d_in[28];
  const float* cb1 = (const float*)d_in[29];
  float* out = (float*)d_out;

  char* ws = (char*)d_ws;
  size_t off = 0;
  auto carve = [&](size_t bytes) -> void* {
    void* p = ws + off;
    off = (off + bytes + 255) & ~(size_t)255;
    return p;
  };
  _Float16* h16  = (_Float16*)carve((size_t)N * CHN * 2);
  _Float16* w16T = (_Float16*)carve((size_t)CHN * CHN * 2);
  float*    hw   = (float*)carve((size_t)N * CHN * 4);
  float*    acc  = (float*)carve((size_t)N * CHN * 4);
  float*    asr  = (float*)carve((size_t)N * HEADS * 4);
  float*    adt  = (float*)carve((size_t)N * HEADS * 4);
  float*    mx   = (float*)carve((size_t)N * HEADS * 4);
  float*    dn   = (float*)carve((size_t)N * HEADS * 4);
  float*    h2   = (float*)carve((size_t)N * HID * 4);
  (void)ws_size; (void)n_in; (void)out_size;

  const int din[3] = {128, CHN, CHN};

  { int n = N * 128; k_cvt16<<<cdiv_i(n, TPB), TPB, 0, stream>>>(x, h16, n); }

  for (int i = 0; i < 3; i++) {
    int K = din[i];
    { int n = K * CHN;
      k_cvtT16<<<cdiv_i(n, TPB), TPB, 0, stream>>>(Wl[i], w16T, K, CHN); }
    { dim3 g(cdiv_i(N, 128), CHN / 64);
      k_gemm_f16_wmma<<<g, TPB, 0, stream>>>(h16, w16T, hw, N, K, CHN); }
    { int n = N * HEADS;
      k_att<<<cdiv_i(n, TPB), TPB, 0, stream>>>(hw, As[i], Ad[i], asr, adt, n); }
    { int n = N * HEADS;
      k_fill<<<cdiv_i(n, TPB), TPB, 0, stream>>>(mx, -1e30f, n); }
    hipMemsetAsync(dn, 0, (size_t)N * HEADS * 4, stream);
    hipMemsetAsync(acc, 0, (size_t)N * CHN * 4, stream);
    { int n = Et * HEADS;
      k_edge_max<<<cdiv_i(n, TPB), TPB, 0, stream>>>(ei, E, Et, asr, adt, mx, n); }
    { int n = Et * HEADS;
      k_edge_denom<<<cdiv_i(n, TPB), TPB, 0, stream>>>(ei, E, Et, asr, adt, mx, dn, n); }
    { int n = Et * 32;
      k_edge_aggr<<<cdiv_i(n, TPB), TPB, 0, stream>>>(ei, E, Et, asr, adt, mx, dn, hw, acc); }
    if (i < 2) {
      int n = N * CHN;
      k_epilogue<<<cdiv_i(n, TPB), TPB, 0, stream>>>(acc, Bi[i], Bg[i], Bb[i], Bm[i], Bv[i], h16, n);
    } else {
      int n = N * HID;
      k_final2<<<cdiv_i(n, TPB), TPB, 0, stream>>>(acc, Bi[2], Bg[2], Bb[2], Bm[2], Bv[2], h2, n);
    }
  }
  k_cls<<<cdiv_i(N, TPB), TPB, 0, stream>>>(h2, cW0, cb0, cW1, cb1, out, N);
}
